// CenterLossLinear_26087631356629
// MI455X (gfx1250) — matrix operand order repaired and checked by reading, not compiled, not run on hardware
//
#include <hip/hip_runtime.h>
#include <stdint.h>

// ---------------------------------------------------------------------------
// CenterLossLinear for MI455X (gfx1250, wave32, WMMA)
//   logits[1024,100000] = embedding[1024,512] @ fc_w[512,100000] + fc_b
//     -> bf16x3 split-precision GEMM on v_wmma_f32_16x16x32_bf16
//   new_centers = centers with scatter-add of -(1-ALPHA)*(centers[l]-emb)
//   center_loss = mean((emb - centers[l])^2)
// Output layout: [logits | center_loss | new_centers] (flat, fp32)
// Workspace: 2 MB = pre-swizzled bf16 hi/lo fragments of embedding.
// ---------------------------------------------------------------------------

typedef __attribute__((ext_vector_type(16))) __bf16 v16bf;
typedef __attribute__((ext_vector_type(8)))  __bf16 v8bf;
typedef __attribute__((ext_vector_type(8)))  float  v8f;

#define ALPHA_C 0.5f
constexpr int GM = 1024;
constexpr int GK = 512;
constexpr int GN = 100000;          // 100000 = 64*1562 + 32
constexpr int NBLK = 64;            // block N tile
constexpr int NBLOCKS = (GN + NBLK - 1) / NBLK;   // 1563

// ---- bf16 split helpers ----------------------------------------------------
__device__ __forceinline__ uint16_t f32_to_bf16(float f) {
  uint32_t u = __float_as_uint(f);
  uint32_t r = (u + 0x7FFFu + ((u >> 16) & 1u)) >> 16;   // round-nearest-even
  return (uint16_t)r;
}
__device__ __forceinline__ float bf16_to_f32(uint16_t h) {
  return __uint_as_float(((uint32_t)h) << 16);
}
// assemble a 32-byte fragment from two naturally-aligned 16B halves
__device__ __forceinline__ v16bf ld_frag16(const uint16_t* p) {
  union { v16bf v; v8bf h[2]; } u;
  u.h[0] = *(const v8bf*)(p);
  u.h[1] = *(const v8bf*)(p + 8);
  return u.v;
}

// ---------------------------------------------------------------------------
// Pre-swizzle embedding into bf16 hi/lo A-fragments (ISA 16-bit A 16x32 layout):
//  lane<16 : V0-3 -> K=2v,2v+1        V4-7 -> K=2v+8..   (16..23)
//  lane>=16: V0-3 -> K=2v+8 (8..15)   V4-7 -> K=2v+16..  (24..31)
// ws u32 index = ((mt*16 + chunk)*32 + lane)*8 + v   (mt = m/16, chunk = k/32)
// => byte offset = mt*16384 + chunk*1024 + lane*32 + v*4
// ---------------------------------------------------------------------------
__global__ __launch_bounds__(256) void preswizzle_A(
    const float* __restrict__ A, uint32_t* __restrict__ ahi,
    uint32_t* __restrict__ alo)
{
  const int idx  = blockIdx.x * 256 + threadIdx.x;     // < 262144
  const int v    = idx & 7;
  const int lane = (idx >> 3) & 31;
  const int c    = (idx >> 8) & 15;
  const int mt   = idx >> 12;
  const int half = lane >> 4, l16 = lane & 15;
  const int m    = mt * 16 + l16;
  const int base = ((v < 4) ? 2 * v : 2 * v + 8) + (half ? 8 : 0);
  const int k    = c * 32 + base;

  const float x0 = A[(size_t)m * GK + k];
  const float x1 = A[(size_t)m * GK + k + 1];
  const uint16_t h0 = f32_to_bf16(x0), h1 = f32_to_bf16(x1);
  const uint16_t l0 = f32_to_bf16(x0 - bf16_to_f32(h0));
  const uint16_t l1 = f32_to_bf16(x1 - bf16_to_f32(h1));
  ahi[idx] = (uint32_t)h0 | ((uint32_t)h1 << 16);
  alo[idx] = (uint32_t)l0 | ((uint32_t)l1 << 16);
}

// ---------------------------------------------------------------------------
// GEMM: grid (1563, 2), block 256 (8 waves = 4 M-groups x 2 N-groups).
// Block tile 512(M) x 64(N); wave tile 128(M) x 32(N) = 8x2 accumulators
// (128 VGPRs acc -> no spills at 2 waves/SIMD).
// fc_w tile (512K x 64N) split-converted once per M-group into LDS (padded
// stride 520 ushorts -> conflict-free ds_load_b128 fragments).
// 3 wmma passes per K=32 chunk: Ahi*Bhi + Ahi*Blo + Alo*Bhi.
// A fragments: SGPR base + 32-bit lane offset (+1024/chunk, s*16384 imm).
// ---------------------------------------------------------------------------
__global__ __launch_bounds__(256) void gemm_logits_bf16x3(
    const uint32_t* __restrict__ ahi, const uint32_t* __restrict__ alo,
    const float* __restrict__ Bm, const float* __restrict__ bias,
    float* __restrict__ C)
{
  constexpr int LDB = 520;                 // 512 + 8 pad (ushorts per column)
  __shared__ uint16_t sBhi[NBLK * LDB];    // [n][k]
  __shared__ uint16_t sBlo[NBLK * LDB];

  const int tid   = threadIdx.x;
  const int lane  = tid & 31, wave = tid >> 5;
  const int l16   = lane & 15, half = lane >> 4;
  const int waveM = wave & 3;              // 4 M groups (128 rows each)
  const int waveN = wave >> 2;             // 2 N groups (32 cols each)
  const int n0    = blockIdx.x * NBLK;
  const int mGrp  = blockIdx.y;            // 2 M block-groups (512 rows each)

  // phase 1: load + split-convert the B column tile
  for (int i = tid; i < NBLK * 512; i += 256) {
    const int n = i & (NBLK - 1), k = i >> 6;
    const int col = n0 + n;
    const float x = (col < GN) ? Bm[(size_t)k * GN + col] : 0.0f;
    const uint16_t h = f32_to_bf16(x);
    sBhi[n * LDB + k] = h;
    sBlo[n * LDB + k] = f32_to_bf16(x - bf16_to_f32(h));
  }
  __syncthreads();

  const v8f vz = {0.f, 0.f, 0.f, 0.f, 0.f, 0.f, 0.f, 0.f};
  v8f acc[8][2];
#pragma unroll
  for (int s = 0; s < 8; ++s) { acc[s][0] = vz; acc[s][1] = vz; }

  const int nW  = waveN * 32;              // wave's column base inside tile
  const int mt0 = mGrp * 32 + waveM * 8;   // 8 m-tiles (128 rows) per wave

  const uint8_t* aH = (const uint8_t*)ahi;
  const uint8_t* aL = (const uint8_t*)alo;
  uint32_t aoff = (uint32_t)(mt0 * 16384 + lane * 32);

  for (int c = 0; c < 16; ++c, aoff += 1024) {   // K chunks of 32
    const int bo = c * 32 + half * 8;      // lane<16: K 0..7,16..23, lane>=16: 8..15,24..31
    auto ldB = [](const uint16_t* p) { union { v16bf v; v8bf h[2]; } u; u.h[0] = *(const v8bf*)(p); u.h[1] = *(const v8bf*)(p + 16); return u.v; };
    const v16bf bh0 = ldB(&sBhi[(nW + l16) * LDB + bo]);
    const v16bf bl0 = ldB(&sBlo[(nW + l16) * LDB + bo]);
    const v16bf bh1 = ldB(&sBhi[(nW + 16 + l16) * LDB + bo]);
    const v16bf bl1 = ldB(&sBlo[(nW + 16 + l16) * LDB + bo]);
#pragma unroll
    for (int s = 0; s < 8; ++s) {
      const v16bf ah = ld_frag16((const uint16_t*)(aH + aoff + s * 16384));
      const v16bf al = ld_frag16((const uint16_t*)(aL + aoff + s * 16384));
      acc[s][0] = __builtin_amdgcn_wmma_f32_16x16x32_bf16(
          false, ah, false, bh0, (short)0, acc[s][0], false, false);
      acc[s][1] = __builtin_amdgcn_wmma_f32_16x16x32_bf16(
          false, ah, false, bh1, (short)0, acc[s][1], false, false);
      acc[s][0] = __builtin_amdgcn_wmma_f32_16x16x32_bf16(
          false, ah, false, bl0, (short)0, acc[s][0], false, false);
      acc[s][1] = __builtin_amdgcn_wmma_f32_16x16x32_bf16(
          false, ah, false, bl1, (short)0, acc[s][1], false, false);
      acc[s][0] = __builtin_amdgcn_wmma_f32_16x16x32_bf16(
          false, al, false, bh0, (short)0, acc[s][0], false, false);
      acc[s][1] = __builtin_amdgcn_wmma_f32_16x16x32_bf16(
          false, al, false, bh1, (short)0, acc[s][1], false, false);
    }
  }

  const int col0 = n0 + nW + l16;
  const int col1 = n0 + nW + 16 + l16;
  const float bv0 = (col0 < GN) ? bias[col0] : 0.0f;
  const float bv1 = (col1 < GN) ? bias[col1] : 0.0f;
  const int rowBase = mGrp * 512 + waveM * 128;
#pragma unroll
  for (int s = 0; s < 8; ++s) {
#pragma unroll
    for (int v = 0; v < 8; ++v) {
      const int row = rowBase + s * 16 + half * 8 + v;  // C: M = vgpr + 8*half
      if (col0 < GN) C[(size_t)row * GN + col0] = acc[s][0][v] + bv0;
      if (col1 < GN) C[(size_t)row * GN + col1] = acc[s][1][v] + bv1;
    }
  }
}

// ---------------------------------------------------------------------------
// centers -> new_centers copy (dst only 4B-aligned in d_out => B32) + loss=0
// ---------------------------------------------------------------------------
__global__ __launch_bounds__(256) void copy_centers(
    const float* __restrict__ src, float* __restrict__ dst,
    float* __restrict__ loss, int n)
{
  const int i = blockIdx.x * 256 + threadIdx.x;
  if (i == 0) *loss = 0.0f;
  if (i < n) dst[i] = src[i];
}

// ---------------------------------------------------------------------------
// gather + scatter-sub (atomics; duplicates accumulate) + loss reduction
// ---------------------------------------------------------------------------
__global__ __launch_bounds__(256) void center_update_loss(
    const float* __restrict__ emb, const int* __restrict__ labels,
    const float* __restrict__ centers, float* __restrict__ newc,
    float* __restrict__ loss)
{
  __shared__ float red[256];
  const int idx = blockIdx.x * 256 + threadIdx.x;   // < GM*GK
  const int b = idx >> 9;                           // D = 512
  const int d = idx & 511;
  const int lab = labels[b];
  const float c = centers[(size_t)lab * 512 + d];
  const float e = emb[idx];
  atomicAdd(&newc[(size_t)lab * 512 + d], -(1.0f - ALPHA_C) * (c - e));
  const float err = e - c;
  red[threadIdx.x] = err * err;
  __syncthreads();
#pragma unroll
  for (int off = 128; off > 0; off >>= 1) {
    if (threadIdx.x < off) red[threadIdx.x] += red[threadIdx.x + off];
    __syncthreads();
  }
  if (threadIdx.x == 0)
    atomicAdd(loss, red[0] * (1.0f / ((float)GM * (float)GK)));
}

// ---------------------------------------------------------------------------
extern "C" void kernel_launch(void* const* d_in, const int* in_sizes, int n_in,
                              void* d_out, int out_size, void* d_ws, size_t ws_size,
                              hipStream_t stream)
{
  (void)in_sizes; (void)n_in; (void)out_size; (void)ws_size;

  const float* emb     = (const float*)d_in[0];   // [1024,512]
  const int*   labels  = (const int*)  d_in[1];   // [1024]
  const float* centers = (const float*)d_in[2];   // [100000,512]
  const float* fc_w    = (const float*)d_in[3];   // [512,100000]
  const float* fc_b    = (const float*)d_in[4];   // [100000]

  float* out    = (float*)d_out;
  float* logits = out;                            // GM*GN
  float* loss   = out + (size_t)GM * GN;          // 1
  float* newc   = loss + 1;                       // GN*GK (U*D)

  uint32_t* ahi = (uint32_t*)d_ws;                // 1 MB
  uint32_t* alo = ahi + (size_t)262144;           // 1 MB

  // 0) split embedding into bf16 hi/lo, pre-swizzled to A-fragment layout
  preswizzle_A<<<262144 / 256, 256, 0, stream>>>(emb, ahi, alo);

  // 1) logits GEMM (bf16x3 on the WMMA pipes), 512M x 64N block tiles
  gemm_logits_bf16x3<<<dim3(NBLOCKS, 2), 256, 0, stream>>>(ahi, alo, fc_w, fc_b, logits);

  // 2) centers copy + loss zero (stream-ordered before scatter)
  const int nCenters = GN * GK;                   // 51,200,000
  copy_centers<<<(nCenters + 255) / 256, 256, 0, stream>>>(centers, newc, loss, nCenters);

  // 3) scatter-sub + loss reduction
  center_update_loss<<<(GM * GK) / 256, 256, 0, stream>>>(emb, labels, centers, newc, loss);
}
